// Transducer_5669356836117
// MI455X (gfx1250) — compile-verified, matrix operand order checked by reading, not run on hardware
//
#include <hip/hip_runtime.h>
#include <hip/hip_bf16.h>

#define F_SZ   80
#define F_PAD  96
#define H_DIM  1024
#define G_DIM  4096
#define OUT_D  512
#define JNT    1024
#define VOCAB  1024
#define B_SZ   4
#define T_LEN  300
#define U_LEN  80

// joint kernel tiling
#define JROWS  64
#define JCOLS  256
#define KCH    64
#define AS_LD  72
#define WS_LD  72

typedef __attribute__((ext_vector_type(16))) _Float16 v16h;
typedef __attribute__((ext_vector_type(8)))  _Float16 v8h;
typedef __attribute__((ext_vector_type(8)))  float    v8f;

// ---------------- fast math (CDNA5 has v_tanh_f32 as a TRANS op) ----------------

__device__ __forceinline__ float fast_tanhf(float x) {
#if __has_builtin(__builtin_amdgcn_tanhf)
    return __builtin_amdgcn_tanhf(x);
#else
    return tanhf(x);
#endif
}

__device__ __forceinline__ float fast_sigmoidf(float x) {
    return 1.f / (1.f + __expf(-x));
}

// ---------------- WMMA helpers (CDNA5 16x16x32 f16 -> f32) ----------------

__device__ __forceinline__ v16h load16(const _Float16* p) {
    // two 16-byte loads: halves 0..7 = K(kb..kb+7), halves 8..15 = K(kb+16..kb+23)
    v8h lo = *(const v8h*)p;
    v8h hi = *(const v8h*)(p + 16);
    v16h r;
#pragma unroll
    for (int j = 0; j < 8; ++j) { r[j] = lo[j]; r[8 + j] = hi[j]; }
    return r;
}

__device__ __forceinline__ v8f wmma_f16(v16h a, v16h b, v8f c) {
    return __builtin_amdgcn_wmma_f32_16x16x32_f16(
        /*neg_a=*/false, a, /*neg_b=*/false, b,
        /*c_mod=*/(short)0, c, /*reuse_a=*/false, /*reuse_b=*/false);
}

// ---------------- generic f16 GEMM:  C[M,N] = A[M,K] @ W[N,K]^T + bias ----------------
// block = 128 threads (4 waves). Block tile 64 rows x 64 cols; wave w -> rows [w*16, w*16+16)
__global__ __launch_bounds__(128)
void gemm_f16_kernel(const _Float16* __restrict__ A, int lda,
                     const _Float16* __restrict__ W, int ldb,
                     const float* __restrict__ bias1,
                     const float* __restrict__ bias2,
                     float* __restrict__ C, _Float16* __restrict__ C16, int ldc,
                     int M, int N, int K)
{
    const int wave = threadIdx.x >> 5;
    const int lane = threadIdx.x & 31;
    const int row0 = blockIdx.y * 64 + wave * 16;
    const int col0 = blockIdx.x * 64;
    if (row0 >= M) return;

    v8f acc[4] = {};
    const int l15   = lane & 15;
    const int kboff = (lane >> 4) << 3;
    const _Float16* Arow = A + (size_t)(row0 + l15) * lda;

    for (int k0 = 0; k0 < K; k0 += 32) {
        const int kb = k0 + kboff;
        v16h af = load16(Arow + kb);
#pragma unroll
        for (int ct = 0; ct < 4; ++ct) {
            const int col = col0 + ct * 16 + l15;
            v16h bf = load16(W + (size_t)col * ldb + kb);
            acc[ct] = wmma_f16(af, bf, acc[ct]);
        }
    }

    const int mb = row0 + ((lane >> 4) << 3);
#pragma unroll
    for (int ct = 0; ct < 4; ++ct) {
        const int col = col0 + ct * 16 + l15;
        float bb = 0.f;
        if (bias1) bb += bias1[col];
        if (bias2) bb += bias2[col];
#pragma unroll
        for (int v = 0; v < 8; ++v) {
            const float val = acc[ct][v] + bb;
            const size_t off = (size_t)(mb + v) * ldc + col;
            C[off] = val;
            if (C16) C16[off] = (_Float16)val;
        }
    }
}

// ---------------- joint kernel v2 ----------------
// out[r, :] = tanh(Aenc[enc(r)] + Bpred[pred(r)]) @ W2^T + b2
// block = 256 threads (8 waves); block tile = 64 rows x 256 cols, K chunked by 64.
// Both the on-the-fly j1 chunk and the W2 chunk are staged in LDS so every
// global W2 byte feeds 64 output rows (4x less L2 traffic than 16-row tiles).
__global__ __launch_bounds__(256)
void joint_kernel(const float* __restrict__ Aenc,   // [B*T, 1024] (includes b1)
                  const float* __restrict__ Bpred,  // [B*U, 1024]
                  const _Float16* __restrict__ W2h, // [1024, 1024]
                  const float* __restrict__ b2,     // [1024]
                  float* __restrict__ out)          // [B*T*U, 1024]
{
    __shared__ alignas(16) _Float16 As[JROWS * AS_LD];  //  9216 B
    __shared__ alignas(16) _Float16 Ws[JCOLS * WS_LD];  // 36864 B
    __shared__ int s_er[JROWS];
    __shared__ int s_pr[JROWS];

    const int row0 = blockIdx.y * JROWS;
    const int colB = blockIdx.x * JCOLS;
    const int tid  = threadIdx.x;

    if (tid < JROWS) {
        const int gr  = row0 + tid;
        const int b   = gr / (T_LEN * U_LEN);
        const int rem = gr % (T_LEN * U_LEN);
        s_er[tid] = b * T_LEN + rem / U_LEN;
        s_pr[tid] = b * U_LEN + rem % U_LEN;
    }

    const int wave  = tid >> 5;
    const int lane  = tid & 31;
    const int l15   = lane & 15;
    const int kboff = (lane >> 4) << 3;
    const int cw    = wave * 32;          // wave's column offset within block tile

    v8f acc[4][2] = {};

    for (int kc = 0; kc < JNT; kc += KCH) {
        __syncthreads();   // previous chunk consumed (also orders s_er init)

        // ---- stage W2 chunk: thread t -> column (colB + t), 64 contiguous halves ----
        {
            const _Float16* src = W2h + (size_t)(colB + tid) * JNT + kc;
            _Float16* dst = &Ws[tid * WS_LD];
#pragma unroll
            for (int j = 0; j < KCH; j += 8)
                *(v8h*)(dst + j) = *(const v8h*)(src + j);
            if (kc + KCH < JNT)
                __builtin_prefetch(src + KCH, 0, 1);   // global_prefetch_b8: next chunk
        }
        // ---- stage j1 chunk: tanh(Aenc + Bpred) -> f16 ----
        for (int idx = tid; idx < JROWS * KCH; idx += 256) {
            const int m  = idx >> 6;          // / KCH
            const int kk = idx & (KCH - 1);
            const float v = Aenc[(size_t)s_er[m] * JNT + kc + kk]
                          + Bpred[(size_t)s_pr[m] * JNT + kc + kk];
            As[m * AS_LD + kk] = (_Float16)fast_tanhf(v);
        }
        __syncthreads();

        // ---- WMMA over the chunk: 4 row subtiles x 2 col subtiles x 2 k-steps ----
#pragma unroll
        for (int ks = 0; ks < KCH; ks += 32) {
            v16h bf0 = load16(&Ws[(cw + l15) * WS_LD + ks + kboff]);
            v16h bf1 = load16(&Ws[(cw + 16 + l15) * WS_LD + ks + kboff]);
#pragma unroll
            for (int r = 0; r < 4; ++r) {
                v16h af = load16(&As[(r * 16 + l15) * AS_LD + ks + kboff]);
                acc[r][0] = wmma_f16(af, bf0, acc[r][0]);
                acc[r][1] = wmma_f16(af, bf1, acc[r][1]);
            }
        }
    }

    // ---- epilogue ----
    const int mb  = (lane >> 4) << 3;
    const int c0g = colB + cw + l15;
    const float bb0 = b2[c0g];
    const float bb1 = b2[c0g + 16];
#pragma unroll
    for (int r = 0; r < 4; ++r) {
        const int rg = row0 + r * 16 + mb;
#pragma unroll
        for (int v = 0; v < 8; ++v) {
            out[(size_t)(rg + v) * VOCAB + c0g]      = acc[r][0][v] + bb0;
            out[(size_t)(rg + v) * VOCAB + c0g + 16] = acc[r][1][v] + bb1;
        }
    }
}

// ---------------- pointwise / prep kernels ----------------

__global__ void cvt2d_kernel(const float* __restrict__ src, int src_ld, int src_cols,
                             _Float16* __restrict__ dst, int dst_ld, int rows)
{
    const int idx = blockIdx.x * blockDim.x + threadIdx.x;
    if (idx >= rows * dst_ld) return;
    const int r = idx / dst_ld;
    const int c = idx % dst_ld;
    dst[idx] = (c < src_cols) ? (_Float16)src[(size_t)r * src_ld + c] : (_Float16)0.f;
}

__global__ void layernorm_kernel(const float* __restrict__ x,
                                 const float* __restrict__ g,
                                 const float* __restrict__ b,
                                 _Float16* __restrict__ out, int rows)
{
    const int r = blockIdx.x * blockDim.x + threadIdx.x;
    if (r >= rows) return;
    const float* xr = x + (size_t)r * F_SZ;
    float mu = 0.f;
    for (int j = 0; j < F_SZ; ++j) mu += xr[j];
    mu *= (1.f / F_SZ);
    float var = 0.f;
    for (int j = 0; j < F_SZ; ++j) { const float d = xr[j] - mu; var += d * d; }
    var *= (1.f / F_SZ);
    const float rs = rsqrtf(var + 1e-5f);
    _Float16* o = out + (size_t)r * F_PAD;
    for (int j = 0; j < F_SZ; ++j) o[j] = (_Float16)((xr[j] - mu) * rs * g[j] + b[j]);
    for (int j = F_SZ; j < F_PAD; ++j) o[j] = (_Float16)0.f;
}

__global__ void lstm_init_kernel(_Float16* __restrict__ h16, float* __restrict__ c)
{
    const int i = blockIdx.x * blockDim.x + threadIdx.x;
    if (i < 16 * H_DIM) h16[i] = (_Float16)0.f;
    if (i < B_SZ * H_DIM) c[i] = 0.f;
}

__global__ void lstm_update_kernel(const float* __restrict__ gx,   // [rows, 4096]
                                   const float* __restrict__ grec, // [16, 4096] rows 0..3 valid
                                   float* __restrict__ c,          // [4, 1024]
                                   _Float16* __restrict__ h16,     // [16, 1024]
                                   float* __restrict__ hseq,       // [rows, 1024]
                                   _Float16* __restrict__ hseq16,  // [rows, 1024]
                                   int t, int seqlen)
{
    const int i = blockIdx.x * blockDim.x + threadIdx.x;
    if (i >= B_SZ * H_DIM) return;
    const int b = i >> 10;
    const int j = i & 1023;
    const size_t grow = (size_t)b * seqlen + t;
    const float* gxr = gx + grow * G_DIM;
    const float* grr = grec + (size_t)b * G_DIM;
    const float gi = gxr[j]        + grr[j];
    const float gf = gxr[1024 + j] + grr[1024 + j];
    const float gg = gxr[2048 + j] + grr[2048 + j];
    const float go = gxr[3072 + j] + grr[3072 + j];
    const float si = fast_sigmoidf(gi);
    const float sf = fast_sigmoidf(gf);
    const float so = fast_sigmoidf(go);
    const float cc = sf * c[i] + si * fast_tanhf(gg);
    const float hh = so * fast_tanhf(cc);
    c[i] = cc;
    h16[(size_t)b * H_DIM + j] = (_Float16)hh;
    hseq[grow * H_DIM + j] = hh;
    hseq16[grow * H_DIM + j] = (_Float16)hh;
}

__global__ void bn_stats_kernel(const float* __restrict__ x, float* __restrict__ mean,
                                float* __restrict__ rstd, int rows)
{
    const int ch = blockIdx.x;
    float s = 0.f, s2 = 0.f;
    for (int r = threadIdx.x; r < rows; r += blockDim.x) {
        const float v = x[(size_t)r * H_DIM + ch];
        s += v; s2 += v * v;
    }
    __shared__ float sh[256], sh2[256];
    sh[threadIdx.x] = s; sh2[threadIdx.x] = s2;
    __syncthreads();
    for (int off = 128; off > 0; off >>= 1) {
        if ((int)threadIdx.x < off) {
            sh[threadIdx.x]  += sh[threadIdx.x + off];
            sh2[threadIdx.x] += sh2[threadIdx.x + off];
        }
        __syncthreads();
    }
    if (threadIdx.x == 0) {
        const float mu  = sh[0] / rows;
        const float var = sh2[0] / rows - mu * mu;
        mean[ch] = mu;
        rstd[ch] = rsqrtf(var + 1e-5f);
    }
}

__global__ void bn_apply_kernel(float* __restrict__ x, _Float16* __restrict__ x16,
                                const float* __restrict__ mean, const float* __restrict__ rstd,
                                const float* __restrict__ g, const float* __restrict__ b,
                                int total)
{
    const int i = blockIdx.x * blockDim.x + threadIdx.x;
    if (i >= total) return;
    const int ch = i & (H_DIM - 1);
    const float v = (x[i] - mean[ch]) * rstd[ch] * g[ch] + b[ch];
    x[i] = v;
    x16[i] = (_Float16)v;
}

__global__ void emb_gather_kernel(const float* __restrict__ emb, const int* __restrict__ y,
                                  _Float16* __restrict__ out, int rows)
{
    const int idx = blockIdx.x * blockDim.x + threadIdx.x;
    if (idx >= rows * H_DIM) return;
    const int r = idx >> 10;
    const int c = idx & 1023;
    out[idx] = (_Float16)emb[(size_t)y[r] * H_DIM + c];
}

// ---------------- host side ----------------

static inline char* ws_take(char*& p, size_t bytes) {
    char* r = p;
    p += (bytes + 255) & ~(size_t)255;
    return r;
}

static inline void cvt2d(const float* src, int src_ld, int src_cols,
                         _Float16* dst, int dst_ld, int rows, hipStream_t s) {
    const int total = rows * dst_ld;
    cvt2d_kernel<<<(total + 255) / 256, 256, 0, s>>>(src, src_ld, src_cols, dst, dst_ld, rows);
}

static inline void gemm(const _Float16* A, int lda, const _Float16* W, int ldb,
                        const float* b1, const float* b2,
                        float* C, _Float16* C16, int ldc,
                        int M, int N, int K, hipStream_t s) {
    dim3 grid(N / 64, (M + 63) / 64);
    gemm_f16_kernel<<<grid, 128, 0, s>>>(A, lda, W, ldb, b1, b2, C, C16, ldc, M, N, K);
}

extern "C" void kernel_launch(void* const* d_in, const int* in_sizes, int n_in,
                              void* d_out, int out_size, void* d_ws, size_t ws_size,
                              hipStream_t stream)
{
    const float* x       = (const float*)d_in[0];
    const int*   y       = (const int*)  d_in[1];
    const float* ln_g    = (const float*)d_in[2];
    const float* ln_b    = (const float*)d_in[3];
    const float* Wih0    = (const float*)d_in[4];
    const float* Whh0    = (const float*)d_in[5];
    const float* bih0    = (const float*)d_in[6];
    const float* bhh0    = (const float*)d_in[7];
    const float* bng0    = (const float*)d_in[10];
    const float* bnb0    = (const float*)d_in[11];
    const float* Wih1    = (const float*)d_in[12];
    const float* Whh1    = (const float*)d_in[13];
    const float* bih1    = (const float*)d_in[14];
    const float* bhh1    = (const float*)d_in[15];
    const float* bng1    = (const float*)d_in[18];
    const float* bnb1    = (const float*)d_in[19];
    const float* enclinW = (const float*)d_in[20];
    const float* enclinB = (const float*)d_in[21];
    const float* emb     = (const float*)d_in[22];
    const float* pWih    = (const float*)d_in[23];
    const float* pWhh    = (const float*)d_in[24];
    const float* pbih    = (const float*)d_in[25];
    const float* pbhh    = (const float*)d_in[26];
    const float* plinW   = (const float*)d_in[27];
    const float* plinB   = (const float*)d_in[28];
    const float* W1      = (const float*)d_in[29];
    const float* b1      = (const float*)d_in[30];
    const float* W2      = (const float*)d_in[31];
    const float* b2      = (const float*)d_in[32];
    float* out = (float*)d_out;

    const int BT = B_SZ * T_LEN;   // 1200
    const int BU = B_SZ * U_LEN;   // 320

    // ---- workspace carve-up ----
    char* p = (char*)d_ws;
    _Float16* ln16      = (_Float16*)ws_take(p, (size_t)BT * F_PAD * 2);
    _Float16* Wih0_16   = (_Float16*)ws_take(p, (size_t)G_DIM * F_PAD * 2);
    _Float16* Whh0_16   = (_Float16*)ws_take(p, (size_t)G_DIM * H_DIM * 2);
    _Float16* Wih1_16   = (_Float16*)ws_take(p, (size_t)G_DIM * H_DIM * 2);
    _Float16* Whh1_16   = (_Float16*)ws_take(p, (size_t)G_DIM * H_DIM * 2);
    _Float16* pWih16    = (_Float16*)ws_take(p, (size_t)G_DIM * H_DIM * 2);
    _Float16* pWhh16    = (_Float16*)ws_take(p, (size_t)G_DIM * H_DIM * 2);
    _Float16* enclinW16 = (_Float16*)ws_take(p, (size_t)OUT_D * H_DIM * 2);
    _Float16* plinW16   = (_Float16*)ws_take(p, (size_t)OUT_D * H_DIM * 2);
    _Float16* W1a16     = (_Float16*)ws_take(p, (size_t)JNT * OUT_D * 2);
    _Float16* W1b16     = (_Float16*)ws_take(p, (size_t)JNT * OUT_D * 2);
    _Float16* W2_16     = (_Float16*)ws_take(p, (size_t)VOCAB * JNT * 2);
    float*    gx        = (float*)   ws_take(p, (size_t)BT * G_DIM * 4);
    float*    grec      = (float*)   ws_take(p, (size_t)16 * G_DIM * 4);
    _Float16* h16       = (_Float16*)ws_take(p, (size_t)16 * H_DIM * 2);
    float*    cstate    = (float*)   ws_take(p, (size_t)B_SZ * H_DIM * 4);
    float*    hseq      = (float*)   ws_take(p, (size_t)BT * H_DIM * 4);
    _Float16* hseq16    = (_Float16*)ws_take(p, (size_t)BT * H_DIM * 2);
    float*    bnmean    = (float*)   ws_take(p, (size_t)H_DIM * 4);
    float*    bnrstd    = (float*)   ws_take(p, (size_t)H_DIM * 4);
    float*    henc      = (float*)   ws_take(p, (size_t)BT * OUT_D * 4);
    _Float16* henc16    = (_Float16*)ws_take(p, (size_t)BT * OUT_D * 2);
    _Float16* emb16     = (_Float16*)ws_take(p, (size_t)BU * H_DIM * 2);
    float*    hpseq     = (float*)   ws_take(p, (size_t)BU * H_DIM * 4);
    _Float16* hpseq16   = (_Float16*)ws_take(p, (size_t)BU * H_DIM * 2);
    float*    hpred     = (float*)   ws_take(p, (size_t)BU * OUT_D * 4);
    _Float16* hpred16   = (_Float16*)ws_take(p, (size_t)BU * OUT_D * 2);
    float*    AencB     = (float*)   ws_take(p, (size_t)BT * JNT * 4);
    float*    BpredB    = (float*)   ws_take(p, (size_t)BU * JNT * 4);
    if ((size_t)(p - (char*)d_ws) > ws_size) return;  // workspace too small

    // ---- weight conversions to f16 ----
    cvt2d(Wih0, F_SZ, F_SZ, Wih0_16, F_PAD, G_DIM, stream);
    cvt2d(Whh0, H_DIM, H_DIM, Whh0_16, H_DIM, G_DIM, stream);
    cvt2d(Wih1, H_DIM, H_DIM, Wih1_16, H_DIM, G_DIM, stream);
    cvt2d(Whh1, H_DIM, H_DIM, Whh1_16, H_DIM, G_DIM, stream);
    cvt2d(pWih, H_DIM, H_DIM, pWih16, H_DIM, G_DIM, stream);
    cvt2d(pWhh, H_DIM, H_DIM, pWhh16, H_DIM, G_DIM, stream);
    cvt2d(enclinW, H_DIM, H_DIM, enclinW16, H_DIM, OUT_D, stream);
    cvt2d(plinW,   H_DIM, H_DIM, plinW16,   H_DIM, OUT_D, stream);
    cvt2d(W1,         2 * OUT_D, OUT_D, W1a16, OUT_D, JNT, stream);   // first half cols
    cvt2d(W1 + OUT_D, 2 * OUT_D, OUT_D, W1b16, OUT_D, JNT, stream);   // second half cols
    cvt2d(W2, JNT, JNT, W2_16, JNT, VOCAB, stream);

    // ---- encoder: LayerNorm -> f16, padded K=96 ----
    layernorm_kernel<<<(BT + 255) / 256, 256, 0, stream>>>(x, ln_g, ln_b, ln16, BT);

    // ---- encoder layer 0 ----
    gemm(ln16, F_PAD, Wih0_16, F_PAD, bih0, bhh0, gx, nullptr, G_DIM, BT, G_DIM, F_PAD, stream);
    lstm_init_kernel<<<(16 * H_DIM + 255) / 256, 256, 0, stream>>>(h16, cstate);
    for (int t = 0; t < T_LEN; ++t) {
        gemm(h16, H_DIM, Whh0_16, H_DIM, nullptr, nullptr, grec, nullptr, G_DIM, 16, G_DIM, H_DIM, stream);
        lstm_update_kernel<<<(B_SZ * H_DIM + 255) / 256, 256, 0, stream>>>(
            gx, grec, cstate, h16, hseq, hseq16, t, T_LEN);
    }
    bn_stats_kernel<<<H_DIM, 256, 0, stream>>>(hseq, bnmean, bnrstd, BT);
    bn_apply_kernel<<<(BT * H_DIM + 255) / 256, 256, 0, stream>>>(
        hseq, hseq16, bnmean, bnrstd, bng0, bnb0, BT * H_DIM);

    // ---- encoder layer 1 ----
    gemm(hseq16, H_DIM, Wih1_16, H_DIM, bih1, bhh1, gx, nullptr, G_DIM, BT, G_DIM, H_DIM, stream);
    lstm_init_kernel<<<(16 * H_DIM + 255) / 256, 256, 0, stream>>>(h16, cstate);
    for (int t = 0; t < T_LEN; ++t) {
        gemm(h16, H_DIM, Whh1_16, H_DIM, nullptr, nullptr, grec, nullptr, G_DIM, 16, G_DIM, H_DIM, stream);
        lstm_update_kernel<<<(B_SZ * H_DIM + 255) / 256, 256, 0, stream>>>(
            gx, grec, cstate, h16, hseq, hseq16, t, T_LEN);
    }
    bn_stats_kernel<<<H_DIM, 256, 0, stream>>>(hseq, bnmean, bnrstd, BT);
    bn_apply_kernel<<<(BT * H_DIM + 255) / 256, 256, 0, stream>>>(
        hseq, hseq16, bnmean, bnrstd, bng1, bnb1, BT * H_DIM);

    // ---- encoder projection (also emits f16 mirror) ----
    gemm(hseq16, H_DIM, enclinW16, H_DIM, enclinB, nullptr, henc, henc16, OUT_D, BT, OUT_D, H_DIM, stream);

    // ---- predictor ----
    emb_gather_kernel<<<(BU * H_DIM + 255) / 256, 256, 0, stream>>>(emb, y, emb16, BU);
    gemm(emb16, H_DIM, pWih16, H_DIM, pbih, pbhh, gx, nullptr, G_DIM, BU, G_DIM, H_DIM, stream);
    lstm_init_kernel<<<(16 * H_DIM + 255) / 256, 256, 0, stream>>>(h16, cstate);
    for (int u = 0; u < U_LEN; ++u) {
        gemm(h16, H_DIM, pWhh16, H_DIM, nullptr, nullptr, grec, nullptr, G_DIM, 16, G_DIM, H_DIM, stream);
        lstm_update_kernel<<<(B_SZ * H_DIM + 255) / 256, 256, 0, stream>>>(
            gx, grec, cstate, h16, hpseq, hpseq16, u, U_LEN);
    }
    gemm(hpseq16, H_DIM, plinW16, H_DIM, plinB, nullptr, hpred, hpred16, OUT_D, BU, OUT_D, H_DIM, stream);

    // ---- joint precompute: Aenc = henc @ W1a^T + b1 ; Bpred = hpred @ W1b^T ----
    gemm(henc16,  OUT_D, W1a16, OUT_D, b1,      nullptr, AencB,  nullptr, JNT, BT, JNT, OUT_D, stream);
    gemm(hpred16, OUT_D, W1b16, OUT_D, nullptr, nullptr, BpredB, nullptr, JNT, BU, JNT, OUT_D, stream);

    // ---- fused joint: tanh(Aenc+Bpred) @ W2^T + b2 over the full B*T*U grid ----
    {
        dim3 grid(VOCAB / JCOLS, (B_SZ * T_LEN * U_LEN) / JROWS);  // (4, 1500)
        joint_kernel<<<grid, 256, 0, stream>>>(AencB, BpredB, W2_16, b2, out);
    }
    (void)in_sizes; (void)n_in; (void)out_size;
}
